// S6_53455162966135
// MI455X (gfx1250) — matrix operand order repaired and checked by reading, not compiled, not run on hardware
//
#include <hip/hip_runtime.h>
#include <hip/hip_bf16.h>
#include <math.h>

// ---------------------------------------------------------------------------
// Problem constants (from the reference)
// ---------------------------------------------------------------------------
#define BB      2
#define LL      2048
#define DMODEL  1024
#define DINNER  2048
#define NSTATE  16
#define DCONV   4
#define DTRANK  128
#define MROWS   (BB * LL)          // 4096 token rows

typedef __bf16 v16bf __attribute__((ext_vector_type(16)));
typedef __bf16 v8bf  __attribute__((ext_vector_type(8)));
typedef float  v8f   __attribute__((ext_vector_type(8)));

// ---------------------------------------------------------------------------
// fp32 -> bf16 conversion (grid-stride)
// ---------------------------------------------------------------------------
__global__ __launch_bounds__(256) void f32_to_bf16_kernel(
    const float* __restrict__ in, __bf16* __restrict__ out, size_t n) {
  size_t i = (size_t)blockIdx.x * blockDim.x + threadIdx.x;
  size_t stride = (size_t)gridDim.x * blockDim.x;
  for (; i < n; i += stride) out[i] = (__bf16)in[i];
}

// ---------------------------------------------------------------------------
// CDNA5 async global -> LDS copy helpers (ASYNCcnt-tracked)
// ---------------------------------------------------------------------------
__device__ __forceinline__ void async_copy_b128(uint32_t lds_byte_addr,
                                                const void* gptr) {
  asm volatile("global_load_async_to_lds_b128 %0, %1, off"
               :
               : "v"(lds_byte_addr), "v"((uint64_t)(uintptr_t)gptr)
               : "memory");
}
__device__ __forceinline__ void wait_async0() {
  asm volatile("s_wait_asynccnt 0x0" ::: "memory");
}

// ---------------------------------------------------------------------------
// LDS-tiled WMMA bf16 GEMM:  C[M,N] = A[M,K] @ W[N,K]^T  (W row-major [N,K])
// Block = 256 threads / 8 waves; tile BM=128 x BN=128 x BK=32, double-buffered
// LDS staged via global_load_async_to_lds_b128. Wave = 32x64 (2x4 WMMA tiles,
// 8 v_wmma per k-chunk per wave). Requires M%128==0, N%128==0, K%32==0.
// EPI: 0 = plain store, 1 = bias + softplus.
// ---------------------------------------------------------------------------
#define LDPAD 40   // padded row stride (elements) in LDS tiles

template <int EPI>
__global__ __launch_bounds__(256) void gemm_bf16_wmma_lds(
    const __bf16* __restrict__ A, int lda,
    const __bf16* __restrict__ W,               // [N, K] row-major
    float* __restrict__ C, int ldc,
    const float* __restrict__ bias,
    int nbn, int K) {
  __shared__ __bf16 As[2][128 * LDPAD];
  __shared__ __bf16 Bs[2][128 * LDPAD];

  const int bm = blockIdx.x / nbn;
  const int bn = blockIdx.x % nbn;
  const int tid  = threadIdx.x;
  const int wave = tid >> 5;
  const int lane = tid & 31;
  const int wm = wave >> 1;                     // 0..3 -> 32-row strip
  const int wn = wave & 1;                      // 0..1 -> 64-col strip
  const int half = lane >> 4;
  const int r    = lane & 15;

  const int m0 = bm * 128;
  const int n0 = bn * 128;

  const uint32_t asb[2] = {(uint32_t)(uintptr_t)&As[0][0],
                           (uint32_t)(uintptr_t)&As[1][0]};
  const uint32_t bsb[2] = {(uint32_t)(uintptr_t)&Bs[0][0],
                           (uint32_t)(uintptr_t)&Bs[1][0]};

  // stage k-chunk k0 into buffer `buf`: A and B tiles are each 128 rows x
  // 32 cols bf16 = 512 x 16B segments; every thread issues 2 + 2 async copies.
  auto stage = [&](int k0, int buf) {
#pragma unroll
    for (int it = 0; it < 2; ++it) {
      const int s   = tid + it * 256;
      const int row = s >> 2;
      const int seg = s & 3;
      async_copy_b128(asb[buf] + (uint32_t)(row * LDPAD + seg * 8) * 2,
                      A + (size_t)(m0 + row) * lda + k0 + seg * 8);
      async_copy_b128(bsb[buf] + (uint32_t)(row * LDPAD + seg * 8) * 2,
                      W + (size_t)(n0 + row) * K + k0 + seg * 8);
    }
  };

  v8f acc[2][4] = {{{}, {}, {}, {}}, {{}, {}, {}, {}}};

  const int KC = K / 32;
  stage(0, 0);
  for (int kc = 0; kc < KC; ++kc) {
    const int cur = kc & 1;
    wait_async0();            // this wave's async copies to buf `cur` landed
    __syncthreads();          // => all waves' copies landed
    if (kc + 1 < KC) stage((kc + 1) * 32, cur ^ 1);

    // fragments from LDS (ISA §7.12.2 layouts)
    const __bf16* as = &As[cur][0];
    const __bf16* bs = &Bs[cur][0];
    v16bf af[2], bfg[4];
#pragma unroll
    for (int i = 0; i < 2; ++i) {
      const __bf16* p = as + (wm * 32 + i * 16 + r) * LDPAD + half * 8;
      v8bf lo = *(const v8bf*)p;
      v8bf hi = *(const v8bf*)(p + 16);
#pragma unroll
      for (int e = 0; e < 8; ++e) { af[i][e] = lo[e]; af[i][8 + e] = hi[e]; }
    }
#pragma unroll
    for (int j = 0; j < 4; ++j) {
      const __bf16* p = bs + (wn * 64 + j * 16 + r) * LDPAD + half * 8;
      v8bf lo = *(const v8bf*)p;
      v8bf hi = *(const v8bf*)(p + 16);
#pragma unroll
      for (int e = 0; e < 8; ++e) { bfg[j][e] = lo[e]; bfg[j][8 + e] = hi[e]; }
    }
#pragma unroll
    for (int i = 0; i < 2; ++i)
#pragma unroll
      for (int j = 0; j < 4; ++j)
        acc[i][j] = __builtin_amdgcn_wmma_f32_16x16x32_bf16(
            false, af[i], false, bfg[j], (short)0, acc[i][j], false, false);

    __syncthreads();          // all reads of buf `cur` done before re-stage
  }

  // store: acc[i][j][v] -> row m0+wm*32+i*16+8*half+v, col n0+wn*64+j*16+r
#pragma unroll
  for (int i = 0; i < 2; ++i) {
#pragma unroll
    for (int j = 0; j < 4; ++j) {
      const int col = n0 + wn * 64 + j * 16 + r;
      float* pc = C + (size_t)(m0 + wm * 32 + i * 16 + half * 8) * ldc + col;
      if (EPI == 0) {
#pragma unroll
        for (int v = 0; v < 8; ++v) pc[(size_t)v * ldc] = acc[i][j][v];
      } else {
        const float bcol = bias[col];
#pragma unroll
        for (int v = 0; v < 8; ++v) {
          float xv = acc[i][j][v] + bcol;
          pc[(size_t)v * ldc] = (xv > 20.f) ? xv : log1pf(__expf(xv));
        }
      }
    }
  }
}

// ---------------------------------------------------------------------------
// Simple wave-per-tile WMMA GEMM (for N=160, not divisible by 128)
// ---------------------------------------------------------------------------
__global__ __launch_bounds__(256) void gemm_bf16_wmma_simple(
    const __bf16* __restrict__ A, int lda,
    const __bf16* __restrict__ W, float* __restrict__ C, int ldc,
    int Mtiles, int Ntiles, int K) {
  const int wave = blockIdx.x * 8 + (threadIdx.x >> 5);
  const int lane = threadIdx.x & 31;
  const int tm = wave / Ntiles;
  const int tn = wave % Ntiles;
  if (tm >= Mtiles) return;
  const int half = lane >> 4;
  const int r    = lane & 15;

  const __bf16* pa = A + (size_t)(tm * 16 + r) * lda + half * 8;
  const __bf16* pw = W + (size_t)(tn * 16 + r) * K + half * 8;

  v8f acc = {};
  for (int k0 = 0; k0 < K; k0 += 32) {
    __builtin_prefetch(pa + 256, 0, 0);
    __builtin_prefetch(pw + 256, 0, 0);
    v8bf a0 = *(const v8bf*)(pa);
    v8bf a1 = *(const v8bf*)(pa + 16);
    v8bf b0 = *(const v8bf*)(pw);
    v8bf b1 = *(const v8bf*)(pw + 16);
    pa += 32;
    pw += 32;
    v16bf av, bv;
#pragma unroll
    for (int i = 0; i < 8; ++i) {
      av[i] = a0[i]; av[8 + i] = a1[i];
      bv[i] = b0[i]; bv[8 + i] = b1[i];
    }
    acc = __builtin_amdgcn_wmma_f32_16x16x32_bf16(
        false, av, false, bv, (short)0, acc, false, false);
  }
  float* pc = C + (size_t)(tm * 16 + half * 8) * ldc + tn * 16 + r;
#pragma unroll
  for (int v = 0; v < 8; ++v) pc[(size_t)v * ldc] = acc[v];
}

// ---------------------------------------------------------------------------
// Causal depthwise conv1d (k=4) + bias + SiLU on xb; SiLU(z) saved for gating.
// ---------------------------------------------------------------------------
__global__ __launch_bounds__(256) void conv_silu_kernel(
    const float* __restrict__ xz, const float* __restrict__ cw,
    const float* __restrict__ cb, float* __restrict__ u,
    __bf16* __restrict__ ubf, float* __restrict__ sz) {
  const size_t idx = (size_t)blockIdx.x * blockDim.x + threadIdx.x;
  if (idx >= (size_t)BB * LL * DINNER) return;
  const int d = (int)(idx % DINNER);
  const int t = (int)((idx / DINNER) % LL);
  const int b = (int)(idx / ((size_t)DINNER * LL));

  const float* base = xz + (size_t)b * LL * (2 * DINNER);
  float s = cb[d];
#pragma unroll
  for (int j = 0; j < DCONV; ++j) {
    const int tt = t - (DCONV - 1) + j;
    if (tt >= 0) s += base[(size_t)tt * (2 * DINNER) + d] * cw[d * DCONV + j];
  }
  const float us = s / (1.f + __expf(-s));
  const size_t o = (size_t)(b * LL + t) * DINNER + d;
  u[o]   = us;
  ubf[o] = (__bf16)us;
  const float z = base[(size_t)t * (2 * DINNER) + DINNER + d];
  sz[o] = z / (1.f + __expf(-z));
}

// ---------------------------------------------------------------------------
// Selective scan. Block = 256 threads = 16 channels x 16 states.
// ---------------------------------------------------------------------------
__global__ __launch_bounds__(256) void scan_kernel(
    const float* __restrict__ u, const float* __restrict__ dt,
    const float* __restrict__ xdbl, const float* __restrict__ Alog,
    const float* __restrict__ Dvec, const float* __restrict__ sz,
    __bf16* __restrict__ ybf) {
  const int b    = blockIdx.x >> 7;
  const int dblk = blockIdx.x & 127;
  const int dl   = threadIdx.x >> 4;
  const int n    = threadIdx.x & 15;
  const int d    = dblk * 16 + dl;

  const float Aan = -__expf(Alog[d * NSTATE + n]);
  const float Dd  = Dvec[d];
  float h = 0.f;

  for (int t = 0; t < LL; ++t) {
    const size_t row = (size_t)(b * LL + t);
    const float ut  = u[row * DINNER + d];
    const float dtt = dt[row * DINNER + d];
    const float Bn  = xdbl[row * (DTRANK + 2 * NSTATE) + DTRANK + n];
    const float Cn  = xdbl[row * (DTRANK + 2 * NSTATE) + DTRANK + NSTATE + n];

    h = __expf(dtt * Aan) * h + (dtt * ut) * Bn;

    float p = h * Cn;
    p += __shfl_xor(p, 8, 16);
    p += __shfl_xor(p, 4, 16);
    p += __shfl_xor(p, 2, 16);
    p += __shfl_xor(p, 1, 16);

    if (n == 0) {
      const float y = (p + ut * Dd) * sz[row * DINNER + d];
      ybf[row * DINNER + d] = (__bf16)y;
    }
  }
}

// ---------------------------------------------------------------------------
// Launch
// ---------------------------------------------------------------------------
extern "C" void kernel_launch(void* const* d_in, const int* in_sizes, int n_in,
                              void* d_out, int out_size, void* d_ws, size_t ws_size,
                              hipStream_t stream) {
  const float* x          = (const float*)d_in[0];  // [B,L,1024]
  const float* in_proj_w  = (const float*)d_in[1];  // [4096,1024]
  const float* conv_w     = (const float*)d_in[2];  // [2048,1,4]
  const float* conv_b     = (const float*)d_in[3];  // [2048]
  const float* x_proj_w   = (const float*)d_in[4];  // [160,2048]
  const float* dt_proj_w  = (const float*)d_in[5];  // [2048,128]
  const float* dt_proj_b  = (const float*)d_in[6];  // [2048]
  const float* A_log      = (const float*)d_in[7];  // [2048,16]
  const float* Dv         = (const float*)d_in[8];  // [2048]
  const float* out_proj_w = (const float*)d_in[9];  // [1024,2048]
  float* out = (float*)d_out;                       // [B,L,1024]

  char* ws = (char*)d_ws;
  size_t o = 0;
  auto take = [&](size_t bytes) { size_t r = o; o += (bytes + 255) & ~(size_t)255; return r; };
  __bf16* w1bf   = (__bf16*)(ws + take((size_t)2 * DINNER * DMODEL * 2));
  __bf16* xbf    = (__bf16*)(ws + take((size_t)MROWS * DMODEL * 2));
  float*  xz     = (float*) (ws + take((size_t)MROWS * 2 * DINNER * 4));
  float*  ubuf   = (float*) (ws + take((size_t)MROWS * DINNER * 4));
  __bf16* ubf    = (__bf16*)(ws + take((size_t)MROWS * DINNER * 2));
  float*  szbuf  = (float*) (ws + take((size_t)MROWS * DINNER * 4));
  __bf16* xpwbf  = (__bf16*)(ws + take((size_t)(DTRANK + 2 * NSTATE) * DINNER * 2));
  float*  xdbl   = (float*) (ws + take((size_t)MROWS * (DTRANK + 2 * NSTATE) * 4));
  __bf16* xdblbf = (__bf16*)(ws + take((size_t)MROWS * (DTRANK + 2 * NSTATE) * 2));
  __bf16* dtwbf  = (__bf16*)(ws + take((size_t)DINNER * DTRANK * 2));
  float*  dtbuf  = (float*) (ws + take((size_t)MROWS * DINNER * 4));
  __bf16* ybf    = (__bf16*)(ws + take((size_t)MROWS * DINNER * 2));
  __bf16* owbf   = (__bf16*)(ws + take((size_t)DMODEL * DINNER * 2));

  // --- bf16 conversions ---
  f32_to_bf16_kernel<<<2048, 256, 0, stream>>>(in_proj_w,  w1bf,  (size_t)2 * DINNER * DMODEL);
  f32_to_bf16_kernel<<<2048, 256, 0, stream>>>(x,          xbf,   (size_t)MROWS * DMODEL);
  f32_to_bf16_kernel<<<512,  256, 0, stream>>>(x_proj_w,   xpwbf, (size_t)(DTRANK + 2 * NSTATE) * DINNER);
  f32_to_bf16_kernel<<<512,  256, 0, stream>>>(dt_proj_w,  dtwbf, (size_t)DINNER * DTRANK);
  f32_to_bf16_kernel<<<2048, 256, 0, stream>>>(out_proj_w, owbf,  (size_t)DMODEL * DINNER);

  // --- 1) xz = x @ in_proj_w^T : M=4096 N=4096 K=1024 (LDS-tiled) ---
  gemm_bf16_wmma_lds<0><<<(MROWS / 128) * ((2 * DINNER) / 128), 256, 0, stream>>>(
      xbf, DMODEL, w1bf, xz, 2 * DINNER, nullptr, (2 * DINNER) / 128, DMODEL);

  // --- 2) causal conv + SiLU, and silu(z) gate ---
  conv_silu_kernel<<<(BB * LL * DINNER) / 256, 256, 0, stream>>>(
      xz, conv_w, conv_b, ubuf, ubf, szbuf);

  // --- 3) x_dbl = u @ x_proj_w^T : M=4096 N=160 K=2048 (simple) ---
  gemm_bf16_wmma_simple<<<(256 * 10) / 8, 256, 0, stream>>>(
      ubf, DINNER, xpwbf, xdbl, DTRANK + 2 * NSTATE,
      MROWS / 16, (DTRANK + 2 * NSTATE) / 16, DINNER);

  f32_to_bf16_kernel<<<1024, 256, 0, stream>>>(
      xdbl, xdblbf, (size_t)MROWS * (DTRANK + 2 * NSTATE));

  // --- 4) dt = softplus(dt_in @ dt_proj_w^T + b) : M=4096 N=2048 K=128 ---
  gemm_bf16_wmma_lds<1><<<(MROWS / 128) * (DINNER / 128), 256, 0, stream>>>(
      xdblbf, DTRANK + 2 * NSTATE, dtwbf, dtbuf, DINNER, dt_proj_b,
      DINNER / 128, DTRANK);

  // --- 5) selective scan (+ D skip + z gate) ---
  scan_kernel<<<BB * (DINNER / 16), 256, 0, stream>>>(
      ubuf, dtbuf, xdbl, A_log, Dv, szbuf, ybf);

  // --- 6) out = y @ out_proj_w^T : M=4096 N=1024 K=2048 (LDS-tiled) ---
  gemm_bf16_wmma_lds<0><<<(MROWS / 128) * (DMODEL / 128), 256, 0, stream>>>(
      ybf, DINNER, owbf, out, DMODEL, nullptr, DMODEL / 128, DINNER);

  (void)in_sizes; (void)n_in; (void)out_size; (void)ws_size;
}
